// QuantisedMonDEQ_4037269258583
// MI455X (gfx1250) — matrix operand order repaired and checked by reading, not compiled, not run on hardware
//
#include <hip/hip_runtime.h>
#include <stdint.h>

// ---------------------------------------------------------------------------
// QuantisedMonDEQ for MI455X (gfx1250, wave32, WMMA)
//
// z <- relu((1-a)z + a(Wq z + Uq x^T + bq)), 40 iterations, then classifier.
// Wq = sW * Wint with Wint integer in [-127,127]  -> exact in f16.
// z carried as f16 hi+lo pair (z = hi + lo), so Wint@z runs on
// v_wmma_f32_16x16x32_f16 (f32 accum) with ~f32 accuracy at f16 rate.
// ---------------------------------------------------------------------------

typedef __attribute__((ext_vector_type(16))) _Float16 v16h;
typedef __attribute__((ext_vector_type(8)))  _Float16 v8h;
typedef __attribute__((ext_vector_type(8)))  float    v8f;

union V16H { v16h v; v8h h[2]; };

#define ALPHA_C 0.5f
#define QMAX_C  127.0f

// problem sizes
#define N_DIM 512     // state dim n
#define D_DIM 784     // input dim d
#define D_PAD 800     // d padded to multiple of 32
#define B_DIM 128     // batch
#define C_DIM 10      // classes
#define ITERS 40

// workspace layout (bytes, all 256-aligned)
#define OFF_ABS   ((size_t)0)            // 3 x u32 abs-max (W, U, b)
#define OFF_BQ    ((size_t)256)          // 512 f32 quantized bias
#define OFF_WF16  ((size_t)2304)         // 512x512 f16 (integer-valued Wint)
#define OFF_UF16  ((size_t)526592)       // 512x800 f16 (integer-valued Uint, padded)
#define OFF_XHI   ((size_t)1345792)      // 128x800 f16 x hi
#define OFF_XLO   ((size_t)1550592)      // 128x800 f16 x lo
#define OFF_UXT   ((size_t)1755392)      // 128x512 f32 (Uq x^T + bq), transposed
#define OFF_ZA    ((size_t)2017536)      // ping: hi at +0, lo at +131072 (each 128x512 f16)
#define OFF_ZB    ((size_t)2279680)      // pong
#define Z_LO_OFF  ((size_t)131072)

// ---------------------------------------------------------------------------
__global__ void k_zero32(uint32_t* p, int n) {
    int i = blockIdx.x * blockDim.x + threadIdx.x;
    if (i < n) p[i] = 0u;
}

// abs-max reduction via integer atomicMax on |f| bit patterns (monotone for >=0)
__global__ void k_absmax(const float* __restrict__ src, int n, unsigned* out) {
    __shared__ unsigned red[256];
    unsigned v = 0u;
    for (int i = blockIdx.x * blockDim.x + threadIdx.x; i < n;
         i += gridDim.x * blockDim.x) {
        unsigned b = __float_as_uint(src[i]) & 0x7fffffffu;
        v = (b > v) ? b : v;
    }
    red[threadIdx.x] = v;
    __syncthreads();
    for (int s = 128; s > 0; s >>= 1) {
        if ((int)threadIdx.x < s) {
            unsigned o = red[threadIdx.x + s];
            if (o > red[threadIdx.x]) red[threadIdx.x] = o;
        }
        __syncthreads();
    }
    if (threadIdx.x == 0) atomicMax(out, red[0]);
}

// Wint[i] = rint(W[i]/sW) stored exactly in f16 (|int| <= 127)
__global__ void k_quant_w(const float* __restrict__ W, const unsigned* am,
                          _Float16* __restrict__ Wf, int n) {
    int i = blockIdx.x * blockDim.x + threadIdx.x;
    if (i >= n) return;
    float s = __uint_as_float(*am) / QMAX_C;
    Wf[i] = (_Float16)rintf(W[i] / s);
}

// Uint padded 512x800 (zeros in k >= 784)
__global__ void k_quant_u(const float* __restrict__ U, const unsigned* am,
                          _Float16* __restrict__ Uf) {
    int i = blockIdx.x * blockDim.x + threadIdx.x;
    if (i >= N_DIM * D_PAD) return;
    int r = i / D_PAD, k = i % D_PAD;
    float s = __uint_as_float(*am) / QMAX_C;
    Uf[i] = (k < D_DIM) ? (_Float16)rintf(U[r * D_DIM + k] / s) : (_Float16)0.0f;
}

__global__ void k_quant_b(const float* __restrict__ b, const unsigned* am,
                          float* __restrict__ bq, int n) {
    int i = blockIdx.x * blockDim.x + threadIdx.x;
    if (i >= n) return;
    float s = __uint_as_float(*am) / QMAX_C;
    bq[i] = rintf(b[i] / s) * s;
}

// x split into f16 hi + lo residual, padded 128x800
__global__ void k_split_x(const float* __restrict__ x,
                          _Float16* __restrict__ xhi, _Float16* __restrict__ xlo) {
    int i = blockIdx.x * blockDim.x + threadIdx.x;
    if (i >= B_DIM * D_PAD) return;
    int r = i / D_PAD, k = i % D_PAD;
    float v = (k < D_DIM) ? x[r * D_DIM + k] : 0.0f;
    _Float16 h = (_Float16)v;
    xhi[i] = h;
    xlo[i] = (_Float16)(v - (float)h);
}

// ---------------------------------------------------------------------------
// UxT[nn][m] = sU * (Uint @ x^T)[m][nn] + bq[m]      (M=512, N=128, K=800)
// One wave per 16x16 output tile. A and B both use contiguous k = 16*hb + s per lane half.
__global__ __launch_bounds__(128) void
k_gemm_ux(const _Float16* __restrict__ Uf, const _Float16* __restrict__ xhi,
          const _Float16* __restrict__ xlo, const unsigned* amU,
          const float* __restrict__ bq, float* __restrict__ UxT) {
    const int lane   = threadIdx.x & 31;
    const int wave   = blockIdx.x * (blockDim.x >> 5) + (threadIdx.x >> 5); // 0..255
    const int tileM  = wave >> 3;       // 0..31
    const int tileN  = wave & 7;        // 0..7
    const int hb     = lane >> 4;       // lane half: 0 or 1
    const int l15    = lane & 15;
    const float sU   = __uint_as_float(*amU) / QMAX_C;

    const _Float16* Arow  = Uf  + (size_t)(tileM * 16 + l15) * D_PAD + hb * 16;
    const _Float16* BhRow = xhi + (size_t)(tileN * 16 + l15) * D_PAD + hb * 16;
    const _Float16* BlRow = xlo + (size_t)(tileN * 16 + l15) * D_PAD + hb * 16;

    v8f acc = {};
    for (int k0 = 0; k0 < D_PAD; k0 += 32) {
        V16H a, bh, bl;
        a.h[0]  = *(const v8h*)(Arow  + k0);        // K = k0+16*hb .. +7
        a.h[1]  = *(const v8h*)(Arow  + k0 + 8);    // K = k0+16*hb+8 .. +7
        bh.h[0] = *(const v8h*)(BhRow + k0);        // K = k0+16*hb .. +7
        bh.h[1] = *(const v8h*)(BhRow + k0 + 8);
        bl.h[0] = *(const v8h*)(BlRow + k0);
        bl.h[1] = *(const v8h*)(BlRow + k0 + 8);
        acc = __builtin_amdgcn_wmma_f32_16x16x32_f16(false, a.v, false, bh.v,
                                                     (short)0, acc, false, false);
        acc = __builtin_amdgcn_wmma_f32_16x16x32_f16(false, a.v, false, bl.v,
                                                     (short)0, acc, false, false);
    }
    const int m0  = tileM * 16 + hb * 8;
    const int nnG = tileN * 16 + l15;
    float* out = UxT + (size_t)nnG * N_DIM + m0;
#pragma unroll
    for (int r = 0; r < 8; ++r) out[r] = sU * acc[r] + bq[m0 + r];
}

// ---------------------------------------------------------------------------
// One fixed-point iteration: z' = relu((1-a)z + a(sW*(Wint@z) + Ux)).
// z stored transposed (128 x 512) as f16 hi/lo so B loads AND the C store
// are contiguous 16B accesses. M=512, N=128, K=512.
__global__ __launch_bounds__(128) void
k_iter(const _Float16* __restrict__ Wf, const unsigned* amW,
       const float* __restrict__ UxT,
       const _Float16* __restrict__ zhiIn, const _Float16* __restrict__ zloIn,
       _Float16* __restrict__ zhiOut, _Float16* __restrict__ zloOut) {
    const int lane  = threadIdx.x & 31;
    const int wave  = blockIdx.x * (blockDim.x >> 5) + (threadIdx.x >> 5);
    const int tileM = wave >> 3;
    const int tileN = wave & 7;
    const int hb    = lane >> 4;
    const int l15   = lane & 15;
    const float sW  = __uint_as_float(*amW) / QMAX_C;

    const _Float16* Arow  = Wf    + (size_t)(tileM * 16 + l15) * N_DIM + hb * 16;
    const _Float16* BhRow = zhiIn + (size_t)(tileN * 16 + l15) * N_DIM + hb * 16;
    const _Float16* BlRow = zloIn + (size_t)(tileN * 16 + l15) * N_DIM + hb * 16;

    v8f acc = {};
#pragma unroll 4
    for (int k0 = 0; k0 < N_DIM; k0 += 32) {
        V16H a, bh, bl;
        a.h[0]  = *(const v8h*)(Arow  + k0);
        a.h[1]  = *(const v8h*)(Arow  + k0 + 8);
        bh.h[0] = *(const v8h*)(BhRow + k0);
        bh.h[1] = *(const v8h*)(BhRow + k0 + 8);
        bl.h[0] = *(const v8h*)(BlRow + k0);
        bl.h[1] = *(const v8h*)(BlRow + k0 + 8);
        acc = __builtin_amdgcn_wmma_f32_16x16x32_f16(false, a.v, false, bh.v,
                                                     (short)0, acc, false, false);
        acc = __builtin_amdgcn_wmma_f32_16x16x32_f16(false, a.v, false, bl.v,
                                                     (short)0, acc, false, false);
    }

    // C-tile element (m, nn): m = tileM*16 + 8*hb + r, nn = tileN*16 + l15
    const int m0  = tileM * 16 + hb * 8;
    const int nnG = tileN * 16 + l15;
    const size_t base = (size_t)nnG * N_DIM + m0;

    v8h zhv = *(const v8h*)(zhiIn + base);
    v8h zlv = *(const v8h*)(zloIn + base);
    v8f uxv = *(const v8f*)(UxT + base);

    v8h oh, ol;
#pragma unroll
    for (int r = 0; r < 8; ++r) {
        float zprev = (float)zhv[r] + (float)zlv[r];
        float u  = (1.0f - ALPHA_C) * zprev + ALPHA_C * (sW * acc[r] + uxv[r]);
        float zn = fmaxf(u, 0.0f);
        _Float16 hh = (_Float16)zn;
        oh[r] = hh;
        ol[r] = (_Float16)(zn - (float)hh);
    }
    *(v8h*)(zhiOut + base) = oh;
    *(v8h*)(zloOut + base) = ol;
}

// ---------------------------------------------------------------------------
// logits[b][c] = dot(z*T[b][:], Wc[c][:]) + bc[c]   (tiny: 1280 dots of 512)
__global__ void k_logits(const _Float16* __restrict__ zhi,
                         const _Float16* __restrict__ zlo,
                         const float* __restrict__ Wc, const float* __restrict__ bc,
                         float* __restrict__ out) {
    int i = blockIdx.x * blockDim.x + threadIdx.x;
    if (i >= B_DIM * C_DIM) return;
    int bb = i / C_DIM, c = i % C_DIM;
    const _Float16* zh = zhi + (size_t)bb * N_DIM;
    const _Float16* zl = zlo + (size_t)bb * N_DIM;
    const float*    wr = Wc  + (size_t)c * N_DIM;
    float s = 0.0f;
    for (int k = 0; k < N_DIM; ++k)
        s += ((float)zh[k] + (float)zl[k]) * wr[k];
    out[i] = s + bc[c];
}

// ---------------------------------------------------------------------------
extern "C" void kernel_launch(void* const* d_in, const int* in_sizes, int n_in,
                              void* d_out, int out_size, void* d_ws, size_t ws_size,
                              hipStream_t stream) {
    (void)in_sizes; (void)n_in; (void)out_size; (void)ws_size;
    const float* W  = (const float*)d_in[0];
    const float* U  = (const float*)d_in[1];
    const float* b  = (const float*)d_in[2];
    const float* x  = (const float*)d_in[3];
    const float* Wc = (const float*)d_in[4];
    const float* bc = (const float*)d_in[5];

    char* ws = (char*)d_ws;
    unsigned*  am   = (unsigned*)(ws + OFF_ABS);
    float*     bq   = (float*)   (ws + OFF_BQ);
    _Float16*  Wf16 = (_Float16*)(ws + OFF_WF16);
    _Float16*  Uf16 = (_Float16*)(ws + OFF_UF16);
    _Float16*  xhi  = (_Float16*)(ws + OFF_XHI);
    _Float16*  xlo  = (_Float16*)(ws + OFF_XLO);
    float*     UxT  = (float*)   (ws + OFF_UXT);
    _Float16*  zhiA = (_Float16*)(ws + OFF_ZA);
    _Float16*  zloA = (_Float16*)(ws + OFF_ZA + Z_LO_OFF);
    _Float16*  zhiB = (_Float16*)(ws + OFF_ZB);
    _Float16*  zloB = (_Float16*)(ws + OFF_ZB + Z_LO_OFF);

    // init: abs-max slots and z0 = 0 (hi+lo of buffer A, contiguous 256KB)
    k_zero32<<<1, 64, 0, stream>>>((uint32_t*)am, 3);
    k_zero32<<<256, 256, 0, stream>>>((uint32_t*)zhiA, 65536);

    // per-tensor abs-max
    k_absmax<<<512, 256, 0, stream>>>(W, N_DIM * N_DIM, am + 0);
    k_absmax<<<784, 256, 0, stream>>>(U, N_DIM * D_DIM, am + 1);
    k_absmax<<<2,   256, 0, stream>>>(b, N_DIM,         am + 2);

    // quantize (weights as exact f16 integers), split x
    k_quant_w<<<1024, 256, 0, stream>>>(W, am + 0, Wf16, N_DIM * N_DIM);
    k_quant_u<<<(N_DIM * D_PAD + 255) / 256, 256, 0, stream>>>(U, am + 1, Uf16);
    k_quant_b<<<2, 256, 0, stream>>>(b, am + 2, bq, N_DIM);
    k_split_x<<<(B_DIM * D_PAD + 255) / 256, 256, 0, stream>>>(x, xhi, xlo);

    // Ux^T = (Uq @ x^T + bq)^T  via WMMA
    k_gemm_ux<<<64, 128, 0, stream>>>(Uf16, xhi, xlo, am + 1, bq, UxT);

    // 40 fixed-point iterations, ping-pong A<->B (even reads A, odd reads B)
    for (int it = 0; it < ITERS; ++it) {
        if ((it & 1) == 0)
            k_iter<<<64, 128, 0, stream>>>(Wf16, am + 0, UxT, zhiA, zloA, zhiB, zloB);
        else
            k_iter<<<64, 128, 0, stream>>>(Wf16, am + 0, UxT, zhiB, zloB, zhiA, zloA);
    }
    // ITERS even -> final z in buffer A

    k_logits<<<(B_DIM * C_DIM + 127) / 128, 128, 0, stream>>>(zhiA, zloA, Wc, bc,
                                                              (float*)d_out);
}
